// EGNNEncoder_38981123178591
// MI455X (gfx1250) — matrix operand order repaired and checked by reading, not compiled, not run on hardware
//
#include <hip/hip_runtime.h>
#include <hip/hip_bf16.h>

typedef __attribute__((ext_vector_type(16))) _Float16 v16h;
typedef __attribute__((ext_vector_type(8)))  float    v8f;

#define H64 64

__device__ __forceinline__ void atomicAddAgent(float* p, float v) {
  __hip_atomic_fetch_add(p, v, __ATOMIC_RELAXED, __HIP_MEMORY_SCOPE_AGENT);
}

__global__ void zero_kernel(float* __restrict__ p, long n) {
  long i = (long)blockIdx.x * blockDim.x + threadIdx.x;
  if (i < n) p[i] = 0.f;
}

// ---------------------------------------------------------------------------
// Edge kernel: per 16-edge wave tile, fused gather -> GEMM1 -> relu -> GEMM2
// -> atomic scatter-add into agg[N,64].  f16 WMMA, f32 accumulate.
// ---------------------------------------------------------------------------
template<int FINL, int KPAD>
__global__ __launch_bounds__(128) void edge_kernel(
    const float* __restrict__ xin, const float* __restrict__ pos,
    const int* __restrict__ row, const int* __restrict__ col,
    const float* __restrict__ ew1, const float* __restrict__ eb1,
    const float* __restrict__ ew2, const float* __restrict__ eb2,
    float* __restrict__ agg, int E)
{
  constexpr int KIN = 2 * FINL + 1;
  __shared__ _Float16 sW1[H64][KPAD];   // ew1 transposed: [n][k], zero-padded K
  __shared__ _Float16 sW2[H64][H64];    // ew2 transposed: [n][k]
  __shared__ float    sB1[H64], sB2[H64];
  __shared__ _Float16 sA[4][16][KPAD];  // per-wave A tile (16 edges x K)
  __shared__ _Float16 sT[4][16][H64];   // per-wave hidden tile
  __shared__ int      sRow[4][16];

  const int tid = threadIdx.x;
  for (int i = tid; i < H64 * KPAD; i += 128) {
    int n = i / KPAD, k = i % KPAD;
    sW1[n][k] = (k < KIN) ? (_Float16)ew1[k * H64 + n] : (_Float16)0.f;
  }
  for (int i = tid; i < H64 * H64; i += 128) {
    int n = i / H64, k = i % H64;
    sW2[n][k] = (_Float16)ew2[k * H64 + n];
  }
  if (tid < H64) { sB1[tid] = eb1[tid]; sB2[tid] = eb2[tid]; }

  const int wave = tid >> 5, lane = tid & 31;
  const long e0 = ((long)blockIdx.x * 4 + wave) * 16;

  if (lane < 16) {
    long e = e0 + lane;
    sRow[wave][lane] = (e < E) ? row[e] : -1;
  }
  // Build A rows: [x_row(F) | x_col(F) | dist], f32 -> f16
  for (int idx = lane; idx < 16 * KIN; idx += 32) {
    int ei = idx / KIN, c = idx % KIN;
    long e = e0 + ei;
    float v = 0.f;
    if (e < E) {
      int r = row[e], cc = col[e];
      if (c < FINL)            v = xin[(long)r * FINL + c];
      else if (c < 2 * FINL)   v = xin[(long)cc * FINL + (c - FINL)];
      else {
        float dx = pos[3*r+0] - pos[3*cc+0];
        float dy = pos[3*r+1] - pos[3*cc+1];
        float dz = pos[3*r+2] - pos[3*cc+2];
        v = sqrtf(dx*dx + dy*dy + dz*dz);
      }
    }
    sA[wave][ei][c] = (_Float16)v;
  }
  constexpr int PADC = KPAD - KIN;
  for (int idx = lane; idx < 16 * PADC; idx += 32) {
    int ei = idx / PADC, c = KIN + idx % PADC;
    sA[wave][ei][c] = (_Float16)0.f;
  }
  __syncthreads();

  const int mrow = lane & 15, hi = lane >> 4, ncol = lane & 15;

  // GEMM1: [16 x KPAD] @ [KPAD x 64]
  v8f acc[4] = {};
  for (int ks = 0; ks < KPAD / 32; ++ks) {
    const int kb = ks * 32;
    v16h a;
    #pragma unroll
    for (int v = 0; v < 8; ++v) {
      const int ka = kb + ((v & 4) ? 16 : 0) + hi * 8 + (v & 3) * 2;
      a[2*v+0] = sA[wave][mrow][ka + 0];
      a[2*v+1] = sA[wave][mrow][ka + 1];
    }
    #pragma unroll
    for (int nt = 0; nt < 4; ++nt) {
      v16h b;
      #pragma unroll
      for (int v = 0; v < 8; ++v) {
        const int kbb = kb + ((v & 4) ? 16 : 0) + hi * 8 + (v & 3) * 2;
        b[2*v+0] = sW1[nt*16 + ncol][kbb + 0];
        b[2*v+1] = sW1[nt*16 + ncol][kbb + 1];
      }
      acc[nt] = __builtin_amdgcn_wmma_f32_16x16x32_f16(
          false, a, false, b, (short)0, acc[nt], false, false);
    }
  }
  // bias + relu -> LDS hidden tile (f16)
  #pragma unroll
  for (int nt = 0; nt < 4; ++nt) {
    const int n = nt*16 + ncol;
    #pragma unroll
    for (int v = 0; v < 8; ++v) {
      const int m = v + hi * 8;
      sT[wave][m][n] = (_Float16)fmaxf(acc[nt][v] + sB1[n], 0.f);
    }
  }
  __syncthreads();

  // GEMM2: [16 x 64] @ [64 x 64]
  v8f acc2[4] = {};
  for (int ks = 0; ks < 2; ++ks) {
    const int kb = ks * 32;
    v16h a;
    #pragma unroll
    for (int v = 0; v < 8; ++v) {
      const int ka = kb + ((v & 4) ? 16 : 0) + hi * 8 + (v & 3) * 2;
      a[2*v+0] = sT[wave][mrow][ka + 0];
      a[2*v+1] = sT[wave][mrow][ka + 1];
    }
    #pragma unroll
    for (int nt = 0; nt < 4; ++nt) {
      v16h b;
      #pragma unroll
      for (int v = 0; v < 8; ++v) {
        const int kbb = kb + ((v & 4) ? 16 : 0) + hi * 8 + (v & 3) * 2;
        b[2*v+0] = sW2[nt*16 + ncol][kbb + 0];
        b[2*v+1] = sW2[nt*16 + ncol][kbb + 1];
      }
      acc2[nt] = __builtin_amdgcn_wmma_f32_16x16x32_f16(
          false, a, false, b, (short)0, acc2[nt], false, false);
    }
  }
  // bias + scatter-add (agg fits in 192MB L2 -> fast agent-scope atomics)
  #pragma unroll
  for (int nt = 0; nt < 4; ++nt) {
    const int n = nt*16 + ncol;
    #pragma unroll
    for (int v = 0; v < 8; ++v) {
      const int m = v + hi * 8;
      const int r = sRow[wave][m];
      if (r >= 0) atomicAddAgent(&agg[(long)r * H64 + n], acc2[nt][v] + sB2[n]);
    }
  }
}

// ---------------------------------------------------------------------------
// Node kernel: out = relu([h|agg]@nw1+nb1)@nw2+nb2 + res, then LayerNorm.
// One thread per node; weights staged in LDS; 64-wide register accumulators.
// ---------------------------------------------------------------------------
template<int FINL, bool HAS_RW>
__global__ __launch_bounds__(256) void node_kernel(
    const float* __restrict__ hin, const float* __restrict__ agg,
    const float* __restrict__ nw1, const float* __restrict__ nb1,
    const float* __restrict__ nw2, const float* __restrict__ nb2,
    const float* __restrict__ gam, const float* __restrict__ bet,
    const float* __restrict__ rw, const float* __restrict__ rb,
    float* __restrict__ hout, int N)
{
  constexpr int K = FINL + H64;
  __shared__ float sW1[K][H64];
  __shared__ float sW2[H64][H64];
  __shared__ float sRW[HAS_RW ? FINL : 1][H64];
  __shared__ float sB1[H64], sB2[H64], sG[H64], sBe[H64], sRB[H64];
  for (int i = threadIdx.x; i < K * H64; i += 256)   sW1[i / H64][i % H64] = nw1[i];
  for (int i = threadIdx.x; i < H64 * H64; i += 256) sW2[i / H64][i % H64] = nw2[i];
  if constexpr (HAS_RW)
    for (int i = threadIdx.x; i < FINL * H64; i += 256) sRW[i / H64][i % H64] = rw[i];
  if (threadIdx.x < H64) {
    sB1[threadIdx.x] = nb1[threadIdx.x];
    sB2[threadIdx.x] = nb2[threadIdx.x];
    sG[threadIdx.x]  = gam[threadIdx.x];
    sBe[threadIdx.x] = bet[threadIdx.x];
    sRB[threadIdx.x] = HAS_RW ? rb[threadIdx.x] : 0.f;
  }
  __syncthreads();
  const long i = (long)blockIdx.x * 256 + threadIdx.x;
  if (i >= N) return;

  float acc[H64];
  #pragma unroll
  for (int n = 0; n < H64; ++n) acc[n] = sB1[n];
  for (int k = 0; k < K; ++k) {
    float v = (k < FINL) ? hin[i * FINL + k] : agg[i * H64 + (k - FINL)];
    #pragma unroll
    for (int n = 0; n < H64; ++n) acc[n] += v * sW1[k][n];
  }
  #pragma unroll
  for (int n = 0; n < H64; ++n) acc[n] = fmaxf(acc[n], 0.f);

  float o[H64];
  #pragma unroll
  for (int n = 0; n < H64; ++n) o[n] = sB2[n];
  for (int k = 0; k < H64; ++k) {
    float v = acc[k];
    #pragma unroll
    for (int n = 0; n < H64; ++n) o[n] += v * sW2[k][n];
  }
  if constexpr (HAS_RW) {
    #pragma unroll
    for (int n = 0; n < H64; ++n) o[n] += sRB[n];
    for (int k = 0; k < FINL; ++k) {
      float v = hin[i * FINL + k];
      #pragma unroll
      for (int n = 0; n < H64; ++n) o[n] += v * sRW[k][n];
    }
  } else {
    #pragma unroll
    for (int n = 0; n < H64; ++n) o[n] += hin[i * H64 + n];
  }
  float mu = 0.f;
  #pragma unroll
  for (int n = 0; n < H64; ++n) mu += o[n];
  mu *= (1.f / H64);
  float var = 0.f;
  #pragma unroll
  for (int n = 0; n < H64; ++n) { float d = o[n] - mu; var += d * d; }
  var *= (1.f / H64);
  const float inv = rsqrtf(var + 1e-5f);
  #pragma unroll
  for (int n = 0; n < H64; ++n)
    hout[i * H64 + n] = (o[n] - mu) * inv * sG[n] + sBe[n];
}

// ---------------------------------------------------------------------------
__global__ __launch_bounds__(256) void pool_kernel(
    const float* __restrict__ h, const int* __restrict__ batch,
    float* __restrict__ pooled, float* __restrict__ counts, int N)
{
  const long i = (long)blockIdx.x * 256 + threadIdx.x;
  if (i >= N) return;
  const int b = batch[i];
  #pragma unroll 8
  for (int n = 0; n < H64; ++n)
    atomicAddAgent(&pooled[(long)b * H64 + n], h[i * H64 + n]);
  atomicAddAgent(&counts[b], 1.f);
}

__global__ void final_kernel(
    const float* __restrict__ pooled, const float* __restrict__ counts,
    const float* __restrict__ w1, const float* __restrict__ b1,
    const float* __restrict__ w2, const float* __restrict__ b2,
    float* __restrict__ out, int B)
{
  const int b = blockIdx.x * blockDim.x + threadIdx.x;
  if (b >= B) return;
  const float inv = 1.f / (counts[b] + 1e-8f);
  float xm[H64];
  #pragma unroll
  for (int n = 0; n < H64; ++n) xm[n] = pooled[b * H64 + n] * inv;
  float t[H64];
  #pragma unroll
  for (int n = 0; n < H64; ++n) t[n] = b1[n];
  for (int k = 0; k < H64; ++k) {
    float v = xm[k];
    #pragma unroll
    for (int n = 0; n < H64; ++n) t[n] += v * w1[k * H64 + n];
  }
  #pragma unroll
  for (int n = 0; n < H64; ++n) t[n] = fmaxf(t[n], 0.f);
  for (int l = 0; l < 32; ++l) {
    float o = b2[l];
    for (int k = 0; k < H64; ++k) o += t[k] * w2[k * 32 + l];
    out[b * 32 + l] = o;
  }
}

// ---------------------------------------------------------------------------
extern "C" void kernel_launch(void* const* d_in, const int* in_sizes, int n_in,
                              void* d_out, int out_size, void* d_ws, size_t ws_size,
                              hipStream_t stream) {
  (void)ws_size;
  const float* x   = (const float*)d_in[0];
  const float* pos = (const float*)d_in[1];
  const int*   ei  = (const int*)d_in[2];
  const int*   bat = (const int*)d_in[3];
  const int N = in_sizes[0] / 16;
  const int E = in_sizes[2] / 2;
  const int B = out_size / 32;
  const int* row = ei;
  const int* col = ei + E;

  const float* P[64];
  for (int i = 0; i < n_in && i < 64; ++i) P[i] = (const float*)d_in[i];

  float* ws     = (float*)d_ws;
  float* hA     = ws;                          // [N,64]
  float* hB     = hA + (size_t)N * 64;         // [N,64]
  float* agg    = hB + (size_t)N * 64;         // [N,64]
  float* pooled = agg + (size_t)N * 64;        // [B,64]
  float* counts = pooled + (size_t)B * 64;     // [B]

  const int  tiles = (E + 15) / 16;
  const int  eblk  = (tiles + 3) / 4;
  const int  nblk  = (N + 255) / 256;
  const long aggN  = (long)N * 64;
  const int  zblk  = (int)((aggN + 255) / 256);

  // Layer 1 (F=16, K=33 -> pad 64; residual via rw)
  zero_kernel<<<zblk, 256, 0, stream>>>(agg, aggN);
  edge_kernel<16, 64><<<eblk, 128, 0, stream>>>(x, pos, row, col,
      P[4], P[5], P[6], P[7], agg, E);
  node_kernel<16, true><<<nblk, 256, 0, stream>>>(x, agg,
      P[8], P[9], P[10], P[11], P[12], P[13], P[34], P[35], hA, N);

  // Layer 2 (F=64, K=129 -> pad 160; identity residual)
  zero_kernel<<<zblk, 256, 0, stream>>>(agg, aggN);
  edge_kernel<64, 160><<<eblk, 128, 0, stream>>>(hA, pos, row, col,
      P[14], P[15], P[16], P[17], agg, E);
  node_kernel<64, false><<<nblk, 256, 0, stream>>>(hA, agg,
      P[18], P[19], P[20], P[21], P[22], P[23], nullptr, nullptr, hB, N);

  // Layer 3
  zero_kernel<<<zblk, 256, 0, stream>>>(agg, aggN);
  edge_kernel<64, 160><<<eblk, 128, 0, stream>>>(hB, pos, row, col,
      P[24], P[25], P[26], P[27], agg, E);
  node_kernel<64, false><<<nblk, 256, 0, stream>>>(hB, agg,
      P[28], P[29], P[30], P[31], P[32], P[33], nullptr, nullptr, hA, N);

  // Mean-pool per graph + head MLP
  const long poolN = (long)B * 64 + B;
  zero_kernel<<<(int)((poolN + 255) / 256), 256, 0, stream>>>(pooled, poolN);
  pool_kernel<<<nblk, 256, 0, stream>>>(hA, bat, pooled, counts, N);
  final_kernel<<<(B + 63) / 64, 64, 0, stream>>>(pooled, counts,
      P[36], P[37], P[38], P[39], (float*)d_out, B);
}
